// ChebyshevConvolutionLin_64364379898206
// MI455X (gfx1250) — compile-verified, matrix operand order checked by reading, not run on hardware
//
#include <hip/hip_runtime.h>
#include <hip/hip_bf16.h>

typedef __attribute__((ext_vector_type(16))) __bf16 v16bf;
typedef __attribute__((ext_vector_type(8)))  float  v8f;
typedef int vint4 __attribute__((vector_size(16)));

#define F_IN 165
#define KP1  192   // F_IN padded to multiple of 32
#define HDIM 512
#define NCLS 2

#define BM 128
#define BN 128
#define BK 32
#define LDT 40     // 32 + 8 pad (bf16 units) -> conflict-free fragment reads

#if defined(__has_builtin)
#if __has_builtin(__builtin_amdgcn_global_load_async_to_lds_b128) && \
    __has_builtin(__builtin_amdgcn_s_wait_asynccnt)
#define USE_ASYNC_LDS 1
#endif
#endif

static __host__ __device__ inline int cdiv(int a, int b) { return (a + b - 1) / b; }

// 16-byte chunk copy global -> LDS (async direct-to-LDS when available)
__device__ __forceinline__ void cp16(const __bf16* g, __bf16* l) {
#ifdef USE_ASYNC_LDS
    __builtin_amdgcn_global_load_async_to_lds_b128(
        (vint4*)g,
        (__attribute__((address_space(3))) vint4*)l, 0, 0);
#else
    *(uint4*)l = *(const uint4*)g;
#endif
}

__device__ __forceinline__ void cp_wait() {
#ifdef USE_ASYNC_LDS
    __builtin_amdgcn_s_wait_asynccnt(0);
#endif
}

// ---------------------------------------------------------------- tiny kernels

__global__ void k_deg(const int* __restrict__ src, float* __restrict__ deg, int E) {
    int i = blockIdx.x * blockDim.x + threadIdx.x;
    if (i < E) atomicAdd(&deg[src[i]], 1.0f);
}

__global__ void k_dis(float* __restrict__ deg, int N) {
    int i = blockIdx.x * blockDim.x + threadIdx.x;
    if (i < N) {
        float d = deg[i];
        deg[i] = (d > 0.0f) ? rsqrtf(d) : 0.0f;   // in place: deg -> dis
    }
}

__global__ void k_wgt(const int* __restrict__ src, const int* __restrict__ dst,
                      const float* __restrict__ dis, float* __restrict__ w, int E) {
    int i = blockIdx.x * blockDim.x + threadIdx.x;
    if (i < E) w[i] = -dis[src[i]] * dis[dst[i]];
}

// fp32 [in_rows x in_cols] -> bf16 [out_rows x out_cols], zero padded
__global__ void k_cvt(const float* __restrict__ in, __bf16* __restrict__ out,
                      int in_rows, int in_cols, int out_rows, int out_cols) {
    long long total = (long long)out_rows * out_cols;
    long long stride = (long long)gridDim.x * blockDim.x;
    for (long long i = blockIdx.x * (long long)blockDim.x + threadIdx.x; i < total; i += stride) {
        int r = (int)(i / out_cols);
        int c = (int)(i % out_cols);
        float v = (r < in_rows && c < in_cols) ? in[(long long)r * in_cols + c] : 0.0f;
        out[i] = (__bf16)v;
    }
}

// weights fp32 [Kin x H] -> bf16 TRANSPOSED [H x Kp], K zero-padded
__global__ void k_cvt_wt(const float* __restrict__ in, __bf16* __restrict__ out,
                         int Kin, int Kp, int H) {
    long long total = (long long)H * Kp;
    long long stride = (long long)gridDim.x * blockDim.x;
    for (long long i = blockIdx.x * (long long)blockDim.x + threadIdx.x; i < total; i += stride) {
        int h = (int)(i / Kp);
        int k = (int)(i % Kp);
        float v = (k < Kin) ? in[(long long)k * H + h] : 0.0f;
        out[i] = (__bf16)v;
    }
}

// P[dst[e], f] += w[e] * x[src[e], f]   (fp32 source)
__global__ void k_prop_f32(const float* __restrict__ x, const int* __restrict__ src,
                           const int* __restrict__ dst, const float* __restrict__ w,
                           float* __restrict__ P, int E, int F, int Pstride) {
    long long total = (long long)E * F;
    long long stride = (long long)gridDim.x * blockDim.x;
    for (long long i = blockIdx.x * (long long)blockDim.x + threadIdx.x; i < total; i += stride) {
        int e = (int)(i / F);
        int f = (int)(i % F);
        float v = w[e] * x[(long long)src[e] * F + f];
        atomicAdd(&P[(long long)dst[e] * Pstride + f], v);
    }
}

// P[dst[e], f] += w[e] * h[src[e], f]   (bf16 source)
__global__ void k_prop_bf16(const __bf16* __restrict__ h, const int* __restrict__ src,
                            const int* __restrict__ dst, const float* __restrict__ w,
                            float* __restrict__ P, int E, int F) {
    long long total = (long long)E * F;
    long long stride = (long long)gridDim.x * blockDim.x;
    for (long long i = blockIdx.x * (long long)blockDim.x + threadIdx.x; i < total; i += stride) {
        int e = (int)(i / F);
        int f = (int)(i % F);
        float v = w[e] * (float)h[(long long)src[e] * F + f];
        atomicAdd(&P[(long long)dst[e] * F + f], v);
    }
}

// ------------------------------------------------- dual-source WMMA bf16 GEMM
// out = relu(A0 @ B0 + A1 @ B1 + bias), out stored bf16.
// A*:  [Nrows x K]  bf16 row-major (K % 32 == 0)
// Bt*: [Hcols x K]  bf16 row-major = TRANSPOSED weights -> B tiles stage like A tiles
__global__ __launch_bounds__(256)
void k_gemm_dual(const __bf16* __restrict__ A0, const __bf16* __restrict__ A1,
                 const __bf16* __restrict__ Bt0, const __bf16* __restrict__ Bt1,
                 const float* __restrict__ bias, __bf16* __restrict__ out,
                 int Nrows, int K, int Hcols) {
    __shared__ __bf16 sA[2][BM * LDT];   // [row][k], double buffered
    __shared__ __bf16 sB[2][BN * LDT];   // [col][k], double buffered

    const int t    = threadIdx.x;
    const int lane = t & 31;
    const int wave = t >> 5;             // 0..7
    const int wm   = wave >> 2;          // 0..1 -> 64-row slab
    const int wn   = wave & 3;           // 0..3 -> 32-col slab
    const int half = lane >> 4;
    const int lp   = lane & 15;

    const int rowBlock = blockIdx.x * BM;
    const int colBlock = blockIdx.y * BN;

    const int ksteps = K / BK;
    const int nsteps = 2 * ksteps;

    v8f acc[4][2];
#pragma unroll
    for (int mt = 0; mt < 4; ++mt)
#pragma unroll
        for (int nt = 0; nt < 2; ++nt)
            acc[mt][nt] = (v8f){0.f, 0.f, 0.f, 0.f, 0.f, 0.f, 0.f, 0.f};

    // issue the 16KB tile-pair copy for `step` into buffer `buf`
    auto issue = [&](int step, int buf) {
        const __bf16* A  = (step < ksteps) ? A0 : A1;
        const __bf16* Bt = (step < ksteps) ? Bt0 : Bt1;
        const int k0 = (step < ksteps ? step : step - ksteps) * BK;
#pragma unroll
        for (int i = 0; i < 2; ++i) {
            int c  = t + i * 256;
            int r  = c >> 2;
            int kc = (c & 3) * 8;
            int gr = rowBlock + r;
            if (gr >= Nrows) gr = Nrows - 1;  // clamp: rows masked at store anyway
            cp16(A + (long long)gr * K + k0 + kc, &sA[buf][r * LDT + kc]);
        }
#pragma unroll
        for (int i = 0; i < 2; ++i) {
            int c  = t + i * 256;
            int r  = c >> 2;
            int kc = (c & 3) * 8;
            cp16(Bt + (long long)(colBlock + r) * K + k0 + kc, &sB[buf][r * LDT + kc]);
        }
    };

    issue(0, 0);

#pragma unroll 1
    for (int step = 0; step < nsteps; ++step) {
        const int buf = step & 1;
        cp_wait();                 // this wave's async copies into sX[buf] done
        __syncthreads();           // all waves done -> tile ready (also fences prior reads)
        if (step + 1 < nsteps) issue(step + 1, buf ^ 1);  // overlap copy with WMMAs

        // fragments per ISA 16-bit layout:
        // lanes 0-15: K {0..7,16..23}; lanes 16-31: K {8..15,24..31}
        v16bf af[4], bfr[2];
#pragma unroll
        for (int mt = 0; mt < 4; ++mt) {
            const __bf16* p = &sA[buf][(wm * 64 + mt * 16 + lp) * LDT];
#pragma unroll
            for (int v = 0; v < 8; ++v) {
                int kb = ((v & 4) ? 16 : 0) + half * 8 + (v & 3) * 2;
                af[mt][2 * v]     = p[kb];
                af[mt][2 * v + 1] = p[kb + 1];
            }
        }
#pragma unroll
        for (int nt = 0; nt < 2; ++nt) {
            const __bf16* p = &sB[buf][(wn * 32 + nt * 16 + lp) * LDT];
#pragma unroll
            for (int v = 0; v < 8; ++v) {
                int kb = ((v & 4) ? 16 : 0) + half * 8 + (v & 3) * 2;
                bfr[nt][2 * v]     = p[kb];
                bfr[nt][2 * v + 1] = p[kb + 1];
            }
        }
#pragma unroll
        for (int mt = 0; mt < 4; ++mt)
#pragma unroll
            for (int nt = 0; nt < 2; ++nt)
                acc[mt][nt] = __builtin_amdgcn_wmma_f32_16x16x32_bf16(
                    false, af[mt], false, bfr[nt], (short)0, acc[mt][nt],
                    false, false);
    }

    // epilogue: bias + relu, bf16 store.  C layout: VGPR r -> M = r + 8*half, N = lp
#pragma unroll
    for (int mt = 0; mt < 4; ++mt) {
#pragma unroll
        for (int nt = 0; nt < 2; ++nt) {
            int gc = colBlock + wn * 32 + nt * 16 + lp;
            float bv = bias[gc];
#pragma unroll
            for (int r = 0; r < 8; ++r) {
                int gr = rowBlock + wm * 64 + mt * 16 + half * 8 + r;
                if (gr < Nrows) {
                    float v = acc[mt][nt][r] + bv;
                    out[(long long)gr * Hcols + gc] = (__bf16)(v > 0.f ? v : 0.f);
                }
            }
        }
    }
}

// ------------------------------------------------------------- final H -> 2
__global__ void k_final(const __bf16* __restrict__ H2, const float* __restrict__ Wl,
                        const float* __restrict__ bl, float* __restrict__ out, int N) {
    int n = blockIdx.x * blockDim.x + threadIdx.x;
    if (n >= N) return;
    float a0 = bl[0], a1 = bl[1];
    const __bf16* h = H2 + (long long)n * HDIM;
#pragma unroll 4
    for (int k = 0; k < HDIM; ++k) {
        float v = (float)h[k];
        a0 += v * Wl[2 * k];
        a1 += v * Wl[2 * k + 1];
    }
    out[2 * n]     = a0;
    out[2 * n + 1] = a1;
}

// -------------------------------------------------------------------- launch
extern "C" void kernel_launch(void* const* d_in, const int* in_sizes, int n_in,
                              void* d_out, int out_size, void* d_ws, size_t ws_size,
                              hipStream_t stream) {
    const float* x    = (const float*)d_in[0];
    const int*   ei   = (const int*)  d_in[1];
    const float* W1_0 = (const float*)d_in[2];
    const float* W1_1 = (const float*)d_in[3];
    const float* b1   = (const float*)d_in[4];
    const float* W2_0 = (const float*)d_in[5];
    const float* W2_1 = (const float*)d_in[6];
    const float* b2   = (const float*)d_in[7];
    const float* Wl   = (const float*)d_in[8];
    const float* bl   = (const float*)d_in[9];
    float* out = (float*)d_out;

    const int N = in_sizes[0] / F_IN;
    const int E = in_sizes[1] / 2;
    const int* src = ei;
    const int* dst = ei + E;

    // workspace carve-out (256B aligned)
    char* p = (char*)d_ws;
    auto take = [&](size_t bytes) {
        char* r = p;
        p += (bytes + 255) & ~(size_t)255;
        return r;
    };
    float*  dis   = (float*) take((size_t)N * 4);            // deg -> dis in place
    float*  wgt   = (float*) take((size_t)E * 4);
    __bf16* Xbf   = (__bf16*)take((size_t)N * KP1 * 2);
    float*  P1f   = (float*) take((size_t)N * KP1 * 4);
    __bf16* P1bf  = (__bf16*)take((size_t)N * KP1 * 2);
    __bf16* W10t  = (__bf16*)take((size_t)HDIM * KP1 * 2);   // transposed [H x Kp]
    __bf16* W11t  = (__bf16*)take((size_t)HDIM * KP1 * 2);
    __bf16* W20t  = (__bf16*)take((size_t)HDIM * HDIM * 2);  // transposed [H x H]
    __bf16* W21t  = (__bf16*)take((size_t)HDIM * HDIM * 2);
    __bf16* H1bf  = (__bf16*)take((size_t)N * HDIM * 2);
    float*  P2f   = (float*) take((size_t)N * HDIM * 4);
    __bf16* P2bf  = (__bf16*)take((size_t)N * HDIM * 2);
    __bf16* H2bf  = (__bf16*)take((size_t)N * HDIM * 2);

    // zero the accumulators (stream-ordered, graph-capturable)
    (void)hipMemsetAsync(dis, 0, (size_t)N * 4, stream);
    (void)hipMemsetAsync(P1f, 0, (size_t)N * KP1 * 4, stream);
    (void)hipMemsetAsync(P2f, 0, (size_t)N * HDIM * 4, stream);

    const int T = 256;
    // edge normalization
    k_deg<<<cdiv(E, T), T, 0, stream>>>(src, dis, E);
    k_dis<<<cdiv(N, T), T, 0, stream>>>(dis, N);
    k_wgt<<<cdiv(E, T), T, 0, stream>>>(src, dst, dis, wgt, E);

    // conversions (pad K to multiple of 32; weights transposed for async staging)
    long long nx = (long long)N * KP1;
    k_cvt<<<(int)((nx + T - 1) / T), T, 0, stream>>>(x, Xbf, N, F_IN, N, KP1);
    k_cvt_wt<<<cdiv(HDIM * KP1, T), T, 0, stream>>>(W1_0, W10t, F_IN, KP1, HDIM);
    k_cvt_wt<<<cdiv(HDIM * KP1, T), T, 0, stream>>>(W1_1, W11t, F_IN, KP1, HDIM);
    k_cvt_wt<<<cdiv(HDIM * HDIM, T), T, 0, stream>>>(W2_0, W20t, HDIM, HDIM, HDIM);
    k_cvt_wt<<<cdiv(HDIM * HDIM, T), T, 0, stream>>>(W2_1, W21t, HDIM, HDIM, HDIM);

    // layer 1: P1 = L_hat x ; H1 = relu(X@W1_0 + P1@W1_1 + b1)
    long long m1 = (long long)E * F_IN;
    k_prop_f32<<<(int)((m1 + T - 1) / T), T, 0, stream>>>(x, src, dst, wgt, P1f, E, F_IN, KP1);
    k_cvt<<<(int)((nx + T - 1) / T), T, 0, stream>>>(P1f, P1bf, N, KP1, N, KP1);
    {
        dim3 grid(cdiv(N, BM), HDIM / BN);
        k_gemm_dual<<<grid, 256, 0, stream>>>(Xbf, P1bf, W10t, W11t, b1, H1bf, N, KP1, HDIM);
    }

    // layer 2: P2 = L_hat h ; H2 = relu(H1@W2_0 + P2@W2_1 + b2)
    long long m2 = (long long)E * HDIM;
    k_prop_bf16<<<(int)((m2 + T - 1) / T), T, 0, stream>>>(H1bf, src, dst, wgt, P2f, E, HDIM);
    long long nh = (long long)N * HDIM;
    k_cvt<<<(int)((nh + T - 1) / T), T, 0, stream>>>(P2f, P2bf, N, HDIM, N, HDIM);
    {
        dim3 grid(cdiv(N, BM), HDIM / BN);
        k_gemm_dual<<<grid, 256, 0, stream>>>(H1bf, P2bf, W20t, W21t, b2, H2bf, N, HDIM, HDIM);
    }

    // final linear -> logits (fp32)
    k_final<<<cdiv(N, T), T, 0, stream>>>(H2bf, Wl, bl, out, N);
}
